// MultiHeadedAttention_15092515078466
// MI455X (gfx1250) — compile-verified, matrix operand order checked by reading, not run on hardware
//
#include <hip/hip_runtime.h>
#include <hip/hip_bf16.h>

typedef _Float16 half_t;
typedef __attribute__((ext_vector_type(16))) _Float16 v16h;
typedef __attribute__((ext_vector_type(8)))  _Float16 v8h;
typedef __attribute__((ext_vector_type(8)))  float    v8f;
typedef __attribute__((ext_vector_type(4)))  int      v4i;

#define B_   2
#define S_   2048
#define D_   1024
#define H_   16
#define DPH_ 64

// ---------------------------------------------------------------------------
// Async global->LDS copy (gfx1250 GLOBAL_LOAD_ASYNC_TO_LDS_B128, ASYNCcnt).
// Builtin signature (from hipcc diagnostic): (v4i AS1*, v4i AS3*, Ii, Ii).
// Falls back to a synchronous v8h copy if the builtin is unavailable.
// ---------------------------------------------------------------------------
#if __has_builtin(__builtin_amdgcn_global_load_async_to_lds_b128)
#define HAVE_ASYNC 1
typedef __attribute__((address_space(1))) v4i gv4i;
typedef __attribute__((address_space(3))) v4i lv4i;
#else
#define HAVE_ASYNC 0
#endif

__device__ __forceinline__ void cp_b128(half_t* lds_dst, const half_t* gsrc) {
#if HAVE_ASYNC
    __builtin_amdgcn_global_load_async_to_lds_b128(
        (gv4i*)const_cast<half_t*>(gsrc), (lv4i*)lds_dst, 0, 0);
#else
    *(v8h*)lds_dst = *(const v8h*)gsrc;
#endif
}

__device__ __forceinline__ void async_wait0() {
#if HAVE_ASYNC
#if __has_builtin(__builtin_amdgcn_s_wait_asynccnt)
    __builtin_amdgcn_s_wait_asynccnt(0);
#else
    asm volatile("s_wait_asynccnt 0" ::: "memory");
#endif
#endif
}

// ---------------------------------------------------------------------------
// A-fragment (and identically-shaped B-fragment) loader for
// v_wmma_f32_16x16x32_f16. Per ISA 7.12.2 (16-bit A 16x32):
//   lane l (row = l&15, hl = l>>4):
//     elements 0..7  = K[kbase + hl*8      .. +7]   (contiguous 16 bytes)
//     elements 8..15 = K[kbase + 16 + hl*8 .. +7]   (contiguous 16 bytes)
// ---------------------------------------------------------------------------
__device__ __forceinline__ v16h frag_a(const half_t* rowptr, int kbase, int hl) {
    union { v16h v; v8h h[2]; } u;
    u.h[0] = *(const v8h*)(rowptr + kbase + hl * 8);
    u.h[1] = *(const v8h*)(rowptr + kbase + 16 + hl * 8);
    return u.v;
}

__device__ __forceinline__ v8f wmma_f16(v16h a, v16h b, v8f c) {
    return __builtin_amdgcn_wmma_f32_16x16x32_f16(false, a, false, b,
                                                  (short)0, c, false, false);
}

// ---------------------------------------------------------------------------
// 128x128-tile GEMM, M=4096 N=1024 K=1024, 256 threads = 8 waves,
// each wave computes 32x64 (2x4 WMMA tiles). f32 (or f16) A, f32 W -> f16 LDS.
// EPI==0: f16 head-major [B,H,S,DPH], value = (acc+bias)*scale   (Q, K)
// EPI==1: f32 row-major  [M,N],       value =  acc+bias          (final out)
// EPI==2: f16 tile-transposed V: [(b*H+h)][s/64][d][s%64]        (V)
// ---------------------------------------------------------------------------
template <int EPI, bool AHALF>
__global__ __launch_bounds__(256) void gemm128(const float* __restrict__ Af,
                                               const half_t* __restrict__ Ah,
                                               const float* __restrict__ W,
                                               const float* __restrict__ bias,
                                               half_t* __restrict__ outH,
                                               float* __restrict__ outF,
                                               float scale) {
    __shared__ __align__(16) half_t As[128][40];  // padded stride: 80B rows
    __shared__ __align__(16) half_t Bs[128][40];  // stored [n][k]

    const int tid  = threadIdx.x;
    const int wid  = tid >> 5;
    const int lane = tid & 31;
    const int l15  = lane & 15;
    const int hl   = lane >> 4;
    const int m0   = blockIdx.y * 128;
    const int n0   = blockIdx.x * 128;
    const int moff = (wid & 3) * 32;   // 4 waves along M
    const int noff = (wid >> 2) * 64;  // 2 waves along N

    v8f acc[2][4];
#pragma unroll
    for (int mt = 0; mt < 2; ++mt)
#pragma unroll
        for (int nt = 0; nt < 4; ++nt) acc[mt][nt] = (v8f){};

    for (int k0 = 0; k0 < D_; k0 += 32) {
        // ---- stage A tile: 128 x 32, convert f32->f16 on the fly ----
        if (AHALF) {
#pragma unroll
            for (int i = 0; i < 2; ++i) {
                int idx = tid + i * 256;                // 512 x v8h
                int row = idx >> 2, c8 = idx & 3;
                *(v8h*)&As[row][c8 * 8] =
                    *(const v8h*)(Ah + (size_t)(m0 + row) * D_ + k0 + c8 * 8);
            }
        } else {
#pragma unroll
            for (int i = 0; i < 4; ++i) {
                int idx = tid + i * 256;                // 1024 x float4
                int row = idx >> 3, c4 = idx & 7;
                float4 f = *(const float4*)(Af + (size_t)(m0 + row) * D_ + k0 + c4 * 4);
                half_t* d = &As[row][c4 * 4];
                d[0] = (half_t)f.x; d[1] = (half_t)f.y;
                d[2] = (half_t)f.z; d[3] = (half_t)f.w;
            }
        }
        // ---- stage B tile transposed: Bs[n][k] <- W[k0+k][n0+n] ----
#pragma unroll
        for (int i = 0; i < 4; ++i) {
            int idx = tid + i * 256;                    // 1024 x float4
            int kr = idx >> 5, c4 = idx & 31;
            float4 f = *(const float4*)(W + (size_t)(k0 + kr) * D_ + n0 + c4 * 4);
            Bs[c4 * 4 + 0][kr] = (half_t)f.x;
            Bs[c4 * 4 + 1][kr] = (half_t)f.y;
            Bs[c4 * 4 + 2][kr] = (half_t)f.z;
            Bs[c4 * 4 + 3][kr] = (half_t)f.w;
        }
        if (k0 + 32 < D_) {  // warm L2/L0 for next K tile
            __builtin_prefetch(AHALF
                                   ? (const void*)(Ah + (size_t)(m0 + (tid >> 1)) * D_ + k0 + 32)
                                   : (const void*)(Af + (size_t)(m0 + (tid >> 1)) * D_ + k0 + 32),
                               0, 0);
            __builtin_prefetch((const void*)(W + (size_t)(k0 + 32 + (tid >> 3)) * D_ + n0), 0, 0);
        }
        __syncthreads();

        v16h af[2], bf[4];
#pragma unroll
        for (int mt = 0; mt < 2; ++mt)
            af[mt] = frag_a(&As[moff + mt * 16 + l15][0], 0, hl);
#pragma unroll
        for (int nt = 0; nt < 4; ++nt)
            bf[nt] = frag_a(&Bs[noff + nt * 16 + l15][0], 0, hl);
#pragma unroll
        for (int mt = 0; mt < 2; ++mt)
#pragma unroll
            for (int nt = 0; nt < 4; ++nt)
                acc[mt][nt] = wmma_f16(af[mt], bf[nt], acc[mt][nt]);
        __syncthreads();
    }

    // ---- epilogue: C/D layout row = r + 8*hl, col = l15 ----
#pragma unroll
    for (int mt = 0; mt < 2; ++mt)
#pragma unroll
        for (int nt = 0; nt < 4; ++nt)
#pragma unroll
            for (int r = 0; r < 8; ++r) {
                int gm = m0 + moff + mt * 16 + 8 * hl + r;
                int gn = n0 + noff + nt * 16 + l15;
                float v = acc[mt][nt][r] + bias[gn];
                if (EPI == 0) {
                    int bb = gm >> 11, s = gm & (S_ - 1);
                    int hh = gn >> 6, d = gn & (DPH_ - 1);
                    outH[(((size_t)(bb * H_ + hh)) * S_ + s) * DPH_ + d] =
                        (half_t)(v * scale);
                } else if (EPI == 2) {
                    int bb = gm >> 11, s = gm & (S_ - 1);
                    int hh = gn >> 6, d = gn & (DPH_ - 1);
                    // tile-transposed V: contiguous along s%64 per (tile, d)
                    outH[(((size_t)(bb * H_ + hh) * (S_ / 64) + (s >> 6)) * DPH_ + d) * 64 +
                         (s & 63)] = (half_t)v;
                } else {
                    outF[(size_t)gm * D_ + gn] = v;
                }
            }
}

// ---------------------------------------------------------------------------
// Fused flash attention: block = 64 queries of one (b,h); 4 waves, each wave
// owns 16 query rows. K/V tiles are double-buffered in LDS and streamed in
// with async global->LDS copies overlapping the WMMA work on the live buffer.
// Online softmax (running max/sum in registers, 16-lane shfl reductions);
// P goes through LDS as f16 to re-layout C-fragments into A-fragments.
// ---------------------------------------------------------------------------
__global__ __launch_bounds__(128) void attn64(const half_t* __restrict__ Qh,
                                              const half_t* __restrict__ Kh,
                                              const half_t* __restrict__ Vts,
                                              const unsigned char* __restrict__ mask,
                                              half_t* __restrict__ ctx) {
    __shared__ __align__(16) half_t Qs[64][72];      // [q][d]
    __shared__ __align__(16) half_t Ks[2][64][72];   // [k][d]  (B-frag for Q*K^T)
    __shared__ __align__(16) half_t Vt[2][64][72];   // [d][k]  (B-frag for P*V)
    __shared__ __align__(16) half_t Ps[64][72];      // [q][k]

    const int tid  = threadIdx.x;
    const int wave = tid >> 5;
    const int lane = tid & 31;
    const int l15  = lane & 15;
    const int hl   = lane >> 4;
    const int q0   = blockIdx.x * 64;
    const int h    = blockIdx.y;
    const int b    = blockIdx.z;
    const size_t headoff = ((size_t)(b * H_ + h)) * S_ * DPH_;
    const size_t vbase   = ((size_t)(b * H_ + h)) * (S_ / 64) * (64 * DPH_);
    const int qr0 = wave * 16;
    const int NT  = S_ / 64;

    auto load_kv = [&](int kt_, int bf) {
        const int k0_ = kt_ * 64;
#pragma unroll
        for (int i = 0; i < 4; ++i) {                // 512 x 16B per matrix
            int idx = tid + i * 128;
            int row = idx >> 3, c8 = idx & 7;
            cp_b128(&Ks[bf][row][c8 * 8],
                    Kh + headoff + (size_t)(k0_ + row) * DPH_ + c8 * 8);
            cp_b128(&Vt[bf][row][c8 * 8],
                    Vts + vbase + (size_t)kt_ * (64 * DPH_) + row * 64 + c8 * 8);
        }
    };

    // ---- prologue: Q tile + first K/V tiles ----
#pragma unroll
    for (int i = 0; i < 4; ++i) {
        int idx = tid + i * 128;
        int row = idx >> 3, c8 = idx & 7;
        cp_b128(&Qs[row][c8 * 8],
                Qh + headoff + (size_t)(q0 + row) * DPH_ + c8 * 8);
    }
    load_kv(0, 0);
    async_wait0();
    __syncthreads();

    v8f o[4];
#pragma unroll
    for (int dt = 0; dt < 4; ++dt) o[dt] = (v8f){};
    float mrow[8], lrow[8];
#pragma unroll
    for (int r = 0; r < 8; ++r) { mrow[r] = -1e30f; lrow[r] = 0.f; }

    int buf = 0;
    for (int kt = 0; kt < NT; ++kt) {
        const int k0 = kt * 64;
        if (kt + 1 < NT) load_kv(kt + 1, buf ^ 1);   // stream next tile async

        // ---- S = Q K^T (Q pre-scaled by 1/sqrt(64)) ----
        v8f s[4];
#pragma unroll
        for (int nt = 0; nt < 4; ++nt) {
            v8f z = (v8f){};
            z = wmma_f16(frag_a(&Qs[qr0 + l15][0], 0, hl),
                         frag_a(&Ks[buf][nt * 16 + l15][0], 0, hl), z);
            z = wmma_f16(frag_a(&Qs[qr0 + l15][0], 32, hl),
                         frag_a(&Ks[buf][nt * 16 + l15][0], 32, hl), z);
            s[nt] = z;
        }

        // ---- mask + per-row max over this tile ----
        float tmax[8];
#pragma unroll
        for (int r = 0; r < 8; ++r) tmax[r] = -1e30f;
        const size_t mbase = ((size_t)b * S_ + (q0 + qr0 + 8 * hl)) * S_ + k0;
#pragma unroll
        for (int nt = 0; nt < 4; ++nt)
#pragma unroll
            for (int r = 0; r < 8; ++r) {
                unsigned char mk = mask[mbase + (size_t)r * S_ + nt * 16 + l15];
                float val = mk ? -1e30f : s[nt][r];
                s[nt][r] = val;
                tmax[r] = fmaxf(tmax[r], val);
            }
#pragma unroll
        for (int off = 1; off < 16; off <<= 1)
#pragma unroll
            for (int r = 0; r < 8; ++r)
                tmax[r] = fmaxf(tmax[r], __shfl_xor(tmax[r], off, 32));

        // ---- online softmax update ----
        float alpha[8], rsum[8];
#pragma unroll
        for (int r = 0; r < 8; ++r) {
            float mn = fmaxf(mrow[r], tmax[r]);
            alpha[r] = __expf(mrow[r] - mn);
            mrow[r]  = mn;
            rsum[r]  = 0.f;
        }
#pragma unroll
        for (int nt = 0; nt < 4; ++nt)
#pragma unroll
            for (int r = 0; r < 8; ++r) {
                float p = __expf(s[nt][r] - mrow[r]);
                rsum[r] += p;
                Ps[qr0 + 8 * hl + r][nt * 16 + l15] = (half_t)p;
            }
#pragma unroll
        for (int off = 1; off < 16; off <<= 1)
#pragma unroll
            for (int r = 0; r < 8; ++r)
                rsum[r] += __shfl_xor(rsum[r], off, 32);
#pragma unroll
        for (int r = 0; r < 8; ++r) lrow[r] = lrow[r] * alpha[r] + rsum[r];
#pragma unroll
        for (int dt = 0; dt < 4; ++dt)
#pragma unroll
            for (int r = 0; r < 8; ++r) o[dt][r] *= alpha[r];

        // ---- O += P V  (Ps rows are wave-private; in-wave LDS is ordered) ----
#pragma unroll
        for (int dt = 0; dt < 4; ++dt) {
            o[dt] = wmma_f16(frag_a(&Ps[qr0 + l15][0], 0, hl),
                             frag_a(&Vt[buf][dt * 16 + l15][0], 0, hl), o[dt]);
            o[dt] = wmma_f16(frag_a(&Ps[qr0 + l15][0], 32, hl),
                             frag_a(&Vt[buf][dt * 16 + l15][0], 32, hl), o[dt]);
        }

        async_wait0();      // next tile fully in LDS (this wave's copies)
        __syncthreads();    // all waves done reading buf / writing buf^1
        buf ^= 1;
    }

    // ---- normalize and write ctx as f16 [B,S,D] (d = h*64 + col) ----
#pragma unroll
    for (int dt = 0; dt < 4; ++dt)
#pragma unroll
        for (int r = 0; r < 8; ++r) {
            int qrow = q0 + qr0 + 8 * hl + r;
            int dcol = h * DPH_ + dt * 16 + l15;
            ctx[((size_t)b * S_ + qrow) * D_ + dcol] =
                (half_t)(o[dt][r] / lrow[r]);
        }
}

// ---------------------------------------------------------------------------
extern "C" void kernel_launch(void* const* d_in, const int* in_sizes, int n_in,
                              void* d_out, int out_size, void* d_ws, size_t ws_size,
                              hipStream_t stream) {
    (void)in_sizes; (void)n_in; (void)out_size; (void)ws_size;
    const float* key   = (const float*)d_in[0];
    const float* value = (const float*)d_in[1];
    const float* query = (const float*)d_in[2];
    const unsigned char* mask = (const unsigned char*)d_in[3];
    const float* Wq = (const float*)d_in[4];
    const float* bq = (const float*)d_in[5];
    const float* Wk = (const float*)d_in[6];
    const float* bk = (const float*)d_in[7];
    const float* Wv = (const float*)d_in[8];
    const float* bv = (const float*)d_in[9];
    const float* Wo = (const float*)d_in[10];
    const float* bo = (const float*)d_in[11];
    float* out = (float*)d_out;

    const size_t headElems = (size_t)B_ * H_ * S_ * DPH_;  // 4M halves each
    half_t* Qh   = (half_t*)d_ws;
    half_t* Kh   = Qh + headElems;
    half_t* Vts  = Kh + headElems;   // tile-transposed V
    half_t* ctxp = Vts + headElems;

    dim3 gg(D_ / 128, (B_ * S_) / 128);
    gemm128<0, false><<<gg, 256, 0, stream>>>(query, nullptr, Wq, bq, Qh, nullptr, 0.125f);
    gemm128<0, false><<<gg, 256, 0, stream>>>(key,   nullptr, Wk, bk, Kh, nullptr, 1.0f);
    gemm128<2, false><<<gg, 256, 0, stream>>>(value, nullptr, Wv, bv, Vts, nullptr, 1.0f);

    dim3 ga(S_ / 64, H_, B_);
    attn64<<<ga, 128, 0, stream>>>(Qh, Kh, Vts, mask, ctxp);

    gemm128<1, true><<<gg, 256, 0, stream>>>(nullptr, ctxp, Wo, bo, nullptr, out, 1.0f);
}
